// MambaBlock_28587302323073
// MI455X (gfx1250) — matrix operand order repaired and checked by reading, not compiled, not run on hardware
//
#include <hip/hip_runtime.h>

// ---------------------------------------------------------------------------
// Mamba block for MI455X (gfx1250, wave32, WMMA).
// GEMMs run on v_wmma_f32_16x16x32_bf16 with double-buffered LDS fed by
// GLOBAL_LOAD_ASYNC_TO_LDS_B128 (ASYNCcnt-tracked, VGPR-bypassing) when the
// builtin is available; scan parallelized over (d_inner x d_state) lanes with
// shfl-xor reduction; conv/gate are fused elementwise.
// ---------------------------------------------------------------------------

typedef __bf16 v16bf __attribute__((ext_vector_type(16)));
typedef float  v8f   __attribute__((ext_vector_type(8)));

#define MB_DMODEL 1024
#define MB_DINNER 2048
#define MB_DSTATE 16
#define MB_DTRANK 64
#define MB_SEQ    2048
#define MB_XZN    (2 * MB_DINNER)              // 4096
#define MB_XDBLN  (MB_DTRANK + 2 * MB_DSTATE)  // 96

// ---- async global->LDS (CDNA5) --------------------------------------------
#if defined(__has_builtin)
#if __has_builtin(__builtin_amdgcn_global_load_async_to_lds_b128)
#define MB_HAVE_ASYNC 1
#endif
#endif

#ifdef MB_HAVE_ASYNC
typedef int mb_v4i __attribute__((ext_vector_type(4)));
typedef __attribute__((address_space(1))) mb_v4i* mb_gptr;  // non-const: builtin param type
typedef __attribute__((address_space(3))) mb_v4i* mb_lptr;
// 16-byte async copy: global -> LDS, bypasses VGPRs, tracked by ASYNCcnt.
__device__ __forceinline__ void mb_async_cp16(const unsigned short* g, unsigned short* l) {
  // AS1 pointer = full VA; AS3 pointer = low 32 bits of the generic shared VA
  // (flat-aperture rule: LDS address is the low 32 bits). Integer round-trips avoid
  // illegal cross-address-space pointer casts (and shed constness).
  __builtin_amdgcn_global_load_async_to_lds_b128(
      (mb_gptr)(unsigned long long)(const void*)g,
      (mb_lptr)(unsigned int)(unsigned long long)(void*)l, 0, 0);
}
#if __has_builtin(__builtin_amdgcn_s_wait_asynccnt)
#define MB_ASYNC_WAIT0() __builtin_amdgcn_s_wait_asynccnt(0)
#else
#define MB_ASYNC_WAIT0() asm volatile("s_wait_asynccnt 0x0" ::: "memory")
#endif
#else
// fallback: synchronous copy through VGPRs
__device__ __forceinline__ void mb_async_cp16(const unsigned short* g, unsigned short* l) {
  *reinterpret_cast<uint4*>(l) = *reinterpret_cast<const uint4*>(g);
}
#define MB_ASYNC_WAIT0() ((void)0)
#endif

__device__ __forceinline__ unsigned short f2bf(float f) {
  unsigned int u = __float_as_uint(f);
  if ((u & 0x7F800000u) == 0x7F800000u) return (unsigned short)(u >> 16); // Inf/NaN
  unsigned int r = u + 0x7FFFu + ((u >> 16) & 1u);                        // RNE
  return (unsigned short)(r >> 16);
}
__device__ __forceinline__ float sigmoidf_fast(float x) { return 1.f / (1.f + __expf(-x)); }
__device__ __forceinline__ float softplusf(float x) { return x > 20.f ? x : log1pf(__expf(x)); }

// ---------------------------------------------------------------------------
// f32 -> bf16 conversion (one-time for weights / activations feeding WMMA)
// ---------------------------------------------------------------------------
__global__ __launch_bounds__(256) void f32_to_bf16_kernel(const float* __restrict__ in,
                                                          unsigned short* __restrict__ out, int n) {
  int i = blockIdx.x * 256 + threadIdx.x;
  if (i < n) out[i] = f2bf(in[i]);
}

// ---------------------------------------------------------------------------
// WMMA GEMM: C(MxN,f32) = A(MxK bf16 row-major) * B(NxK bf16 row-major)^T
//  MODE 0: outF[m*N+n] = acc
//  MODE 1: outF[n*M+m] = softplus(acc + bias[n])            (transposed, dt path)
//  MODE 2: n<64 -> outBF[m*64+n] = bf16(acc)  (dt_in operand)
//          n>=64 -> outBC[m*32+(n-64)] = acc  (B/C for the scan)
// Double-buffered LDS; tile k+1 streams in via async-to-LDS while tile k runs
// on the WMMA pipe. Fragment layouts per CDNA5 ISA §7.12.2.
// ---------------------------------------------------------------------------
union FragBF { uint4 u[2]; v16bf v; };

template <int BM, int BN, int WMT, int WNT, int WAVES_M, int WAVES_N, int MODE>
__global__ __launch_bounds__(256) void wmma_gemm_kernel(
    const unsigned short* __restrict__ A, const unsigned short* __restrict__ B,
    float* __restrict__ outF, unsigned short* __restrict__ outBF,
    float* __restrict__ outBC, const float* __restrict__ bias, int M, int N, int K) {
  static_assert(BM == WAVES_M * WMT * 16 && BN == WAVES_N * WNT * 16, "tile mismatch");
  static_assert(WAVES_M * WAVES_N * 32 == 256, "block must be 8 waves");
  constexpr int KSTEP = 32;
  constexpr int KP = 40;  // padded LDS row stride in bf16 elements (80B, 16B-aligned)
  __shared__ __align__(16) unsigned short As[2][BM * KP];
  __shared__ __align__(16) unsigned short Bs[2][BN * KP];

  const int tid = threadIdx.x;
  const int lane = tid & 31;
  const int lr = lane & 15;   // row (A) / col (B) / N (C)
  const int kh = lane >> 4;   // lane-half selects K group / M-half of C
  const int wave = tid >> 5;
  const int m_wave = (wave % WAVES_M) * WMT * 16;
  const int n_wave = (wave / WAVES_M) * WNT * 16;
  const int m0 = blockIdx.y * BM;
  const int n0 = blockIdx.x * BN;

  // stage one KSTEP-slab of A and B into LDS buffer `buf` (async)
  auto stage = [&](int buf, int k0) {
    for (int idx = tid; idx < BM * 2; idx += 256) {
      int row = idx >> 1, half = idx & 1;
      const unsigned short* g = A + (size_t)(m0 + row) * K + k0 + half * 16;
      unsigned short* l = &As[buf][row * KP + half * 16];
      mb_async_cp16(g, l);
      mb_async_cp16(g + 8, l + 8);
    }
    for (int idx = tid; idx < BN * 2; idx += 256) {
      int row = idx >> 1, half = idx & 1;
      const unsigned short* g = B + (size_t)(n0 + row) * K + k0 + half * 16;
      unsigned short* l = &Bs[buf][row * KP + half * 16];
      mb_async_cp16(g, l);
      mb_async_cp16(g + 8, l + 8);
    }
  };

  v8f acc[WMT][WNT];
  v8f vzero = {};
#pragma unroll
  for (int mi = 0; mi < WMT; ++mi)
#pragma unroll
    for (int ni = 0; ni < WNT; ++ni) acc[mi][ni] = vzero;

  stage(0, 0);
  int buf = 0;
  for (int k0 = 0; k0 < K; k0 += KSTEP) {
    // Only the current tile's async copies are outstanding per wave here, so a
    // uniform wait-to-zero is exact; barrier then makes all writes visible and
    // guarantees every wave is done reading the *other* buffer (its ds_load
    // fragments completed before its previous WMMA, which precedes arrival).
    MB_ASYNC_WAIT0();
    __syncthreads();

    FragBF afrag[WMT], bfrag[WNT];
#pragma unroll
    for (int mi = 0; mi < WMT; ++mi) {
      int r = m_wave + mi * 16 + lr;
      // A 16x32: this lane-half holds K = kh*8+{0..7} and 16+kh*8+{0..7}
      afrag[mi].u[0] = *reinterpret_cast<const uint4*>(&As[buf][r * KP + kh * 8]);
      afrag[mi].u[1] = *reinterpret_cast<const uint4*>(&As[buf][r * KP + 16 + kh * 8]);
    }
#pragma unroll
    for (int ni = 0; ni < WNT; ++ni) {
      int r = n_wave + ni * 16 + lr;
      // B 32x16: this lane-half holds K = kh*8+{0..7} and 16+kh*8+{0..7}
      bfrag[ni].u[0] = *reinterpret_cast<const uint4*>(&Bs[buf][r * KP + kh * 8]);
      bfrag[ni].u[1] = *reinterpret_cast<const uint4*>(&Bs[buf][r * KP + 16 + kh * 8]);
    }

    // prefetch next slab into the other buffer while this one computes
    if (k0 + KSTEP < K) stage(buf ^ 1, k0 + KSTEP);

#pragma unroll
    for (int mi = 0; mi < WMT; ++mi)
#pragma unroll
      for (int ni = 0; ni < WNT; ++ni)
        acc[mi][ni] = __builtin_amdgcn_wmma_f32_16x16x32_bf16(
            false, afrag[mi].v, false, bfrag[ni].v, (short)0, acc[mi][ni], false, false);

    buf ^= 1;
  }

  // epilogue: C/D layout -> m = kh*8 + r, n = lane&15
#pragma unroll
  for (int mi = 0; mi < WMT; ++mi) {
#pragma unroll
    for (int ni = 0; ni < WNT; ++ni) {
      int gm_base = m0 + m_wave + mi * 16 + kh * 8;
      int gn = n0 + n_wave + ni * 16 + lr;
#pragma unroll
      for (int r = 0; r < 8; ++r) {
        int gm = gm_base + r;
        float v = acc[mi][ni][r];
        if (MODE == 0) {
          outF[(size_t)gm * N + gn] = v;
        } else if (MODE == 1) {
          outF[(size_t)gn * M + gm] = softplusf(v + bias[gn]);
        } else {
          if (gn < MB_DTRANK) outBF[(size_t)gm * MB_DTRANK + gn] = f2bf(v);
          else                outBC[(size_t)gm * 32 + (gn - MB_DTRANK)] = v;
        }
      }
    }
  }
}

// ---------------------------------------------------------------------------
// Depthwise causal conv (4-tap) + SiLU. Emits xs in bf16 (L,d) for the x_dbl
// GEMM and in f32 transposed (d,L) for contiguous scan streaming (LDS tile
// transpose keeps both writes coalesced).
// ---------------------------------------------------------------------------
__global__ __launch_bounds__(256) void conv_silu_kernel(const float* __restrict__ xz,
                                                        const float* __restrict__ w,
                                                        const float* __restrict__ b,
                                                        unsigned short* __restrict__ xs_bf,
                                                        float* __restrict__ xsT) {
  __shared__ float tile[32][33];
  const int tx = threadIdx.x;  // d within tile
  const int ty = threadIdx.y;  // 0..7
  const int d0 = blockIdx.x * 32;
  const int l0 = blockIdx.y * 32;
  const int d = d0 + tx;
  const float w0 = w[d * 4 + 0], w1 = w[d * 4 + 1], w2 = w[d * 4 + 2], w3 = w[d * 4 + 3];
  const float bb = b[d];
#pragma unroll
  for (int i = 0; i < 4; ++i) {
    int ll = ty + i * 8;
    int l = l0 + ll;
    float acc = bb + w3 * xz[(size_t)l * MB_XZN + d];
    if (l - 1 >= 0) acc += w2 * xz[(size_t)(l - 1) * MB_XZN + d];
    if (l - 2 >= 0) acc += w1 * xz[(size_t)(l - 2) * MB_XZN + d];
    if (l - 3 >= 0) acc += w0 * xz[(size_t)(l - 3) * MB_XZN + d];
    float s = acc * sigmoidf_fast(acc);  // SiLU
    xs_bf[(size_t)l * MB_DINNER + d] = f2bf(s);
    tile[ll][tx] = s;
  }
  __syncthreads();
#pragma unroll
  for (int i = 0; i < 4; ++i) {
    int dl = ty + i * 8;
    xsT[(size_t)(d0 + dl) * MB_SEQ + (l0 + tx)] = tile[tx][dl];
  }
}

// ---------------------------------------------------------------------------
// Selective scan: one lane per (channel d, state n). 16-lane half-wave owns a
// channel; y_t = sum_n h*C via shfl-xor reduction. Serial only in t.
// ---------------------------------------------------------------------------
__global__ __launch_bounds__(256) void scan_kernel(const float* __restrict__ dtT,
                                                   const float* __restrict__ xsT,
                                                   const float* __restrict__ bc,
                                                   const float* __restrict__ A_log,
                                                   const float* __restrict__ Dvec,
                                                   float* __restrict__ y) {
  const int tid = threadIdx.x;
  const int wave = tid >> 5;
  const int lane = tid & 31;
  const int n = lane & 15;
  const int ch = blockIdx.x * 16 + wave * 2 + (lane >> 4);
  const float Acoef = -__expf(A_log[ch * MB_DSTATE + n]);
  const float Dd = Dvec[ch];
  const float* dtrow = dtT + (size_t)ch * MB_SEQ;
  const float* xsrow = xsT + (size_t)ch * MB_SEQ;
  float h = 0.f;
  for (int t = 0; t < MB_SEQ; ++t) {
    float dt = dtrow[t];
    float xv = xsrow[t];
    float Bn = bc[t * 32 + n];
    float Cn = bc[t * 32 + 16 + n];
    float dA = __expf(dt * Acoef);
    h = fmaf(dA, h, dt * Bn * xv);
    float c = h * Cn;
    c += __shfl_xor(c, 1);
    c += __shfl_xor(c, 2);
    c += __shfl_xor(c, 4);
    c += __shfl_xor(c, 8);
    if (n == 0) y[(size_t)t * MB_DINNER + ch] = c + Dd * xv;
  }
}

// ---------------------------------------------------------------------------
// Gate: yg = bf16( y * silu(z) ), z from the second half of xz.
// ---------------------------------------------------------------------------
__global__ __launch_bounds__(256) void gate_silu_kernel(const float* __restrict__ y,
                                                        const float* __restrict__ xz,
                                                        unsigned short* __restrict__ yg) {
  size_t i = (size_t)blockIdx.x * 256 + threadIdx.x;
  int l = (int)(i / MB_DINNER);
  int d = (int)(i % MB_DINNER);
  float z = xz[(size_t)l * MB_XZN + MB_DINNER + d];
  yg[i] = f2bf(y[i] * z * sigmoidf_fast(z));
}

// ---------------------------------------------------------------------------
extern "C" void kernel_launch(void* const* d_in, const int* in_sizes, int n_in,
                              void* d_out, int out_size, void* d_ws, size_t ws_size,
                              hipStream_t stream) {
  const float* x      = (const float*)d_in[0];
  const float* W_in   = (const float*)d_in[1];
  const float* conv_w = (const float*)d_in[2];
  const float* conv_b = (const float*)d_in[3];
  const float* W_x    = (const float*)d_in[4];
  const float* W_dt   = (const float*)d_in[5];
  const float* b_dt   = (const float*)d_in[6];
  const float* A_log  = (const float*)d_in[7];
  const float* Dv     = (const float*)d_in[8];
  const float* W_out  = (const float*)d_in[9];
  (void)in_sizes; (void)n_in; (void)out_size; (void)ws_size;

  char* p = (char*)d_ws;
  auto carve = [&](size_t bytes) {
    char* q = p;
    p += (bytes + 255) & ~(size_t)255;
    return (void*)q;
  };
  unsigned short* x_bf    = (unsigned short*)carve((size_t)MB_SEQ * MB_DMODEL * 2);
  unsigned short* Win_bf  = (unsigned short*)carve((size_t)MB_XZN * MB_DMODEL * 2);
  unsigned short* Wx_bf   = (unsigned short*)carve((size_t)MB_XDBLN * MB_DINNER * 2);
  unsigned short* Wdt_bf  = (unsigned short*)carve((size_t)MB_DINNER * MB_DTRANK * 2);
  unsigned short* Wout_bf = (unsigned short*)carve((size_t)MB_DMODEL * MB_DINNER * 2);
  float*          xz      = (float*)carve((size_t)MB_SEQ * MB_XZN * 4);
  unsigned short* xs_bf   = (unsigned short*)carve((size_t)MB_SEQ * MB_DINNER * 2);
  float*          xsT     = (float*)carve((size_t)MB_DINNER * MB_SEQ * 4);
  unsigned short* dtin_bf = (unsigned short*)carve((size_t)MB_SEQ * MB_DTRANK * 2);
  float*          bc      = (float*)carve((size_t)MB_SEQ * 32 * 4);
  float*          dtT     = (float*)carve((size_t)MB_DINNER * MB_SEQ * 4);
  float*          y       = (float*)carve((size_t)MB_SEQ * MB_DINNER * 4);
  unsigned short* yg_bf   = (unsigned short*)carve((size_t)MB_SEQ * MB_DINNER * 2);

  auto cvt = [&](const float* src, unsigned short* dst, int n) {
    f32_to_bf16_kernel<<<(n + 255) / 256, 256, 0, stream>>>(src, dst, n);
  };
  cvt(x, x_bf, MB_SEQ * MB_DMODEL);
  cvt(W_in, Win_bf, MB_XZN * MB_DMODEL);
  cvt(W_x, Wx_bf, MB_XDBLN * MB_DINNER);
  cvt(W_dt, Wdt_bf, MB_DINNER * MB_DTRANK);
  cvt(W_out, Wout_bf, MB_DMODEL * MB_DINNER);

  // xz = x @ W_in^T : (2048x1024)(4096x1024)^T
  wmma_gemm_kernel<128, 128, 2, 4, 4, 2, 0>
      <<<dim3(MB_XZN / 128, MB_SEQ / 128), 256, 0, stream>>>(
          x_bf, Win_bf, xz, nullptr, nullptr, nullptr, MB_SEQ, MB_XZN, MB_DMODEL);

  conv_silu_kernel<<<dim3(MB_DINNER / 32, MB_SEQ / 32), dim3(32, 8), 0, stream>>>(
      xz, conv_w, conv_b, xs_bf, xsT);

  // x_dbl = xs @ W_x^T : N=96 -> dt_in (bf16) + B/C (f32)
  wmma_gemm_kernel<128, 32, 1, 2, 8, 1, 2>
      <<<dim3(MB_XDBLN / 32, MB_SEQ / 128), 256, 0, stream>>>(
          xs_bf, Wx_bf, nullptr, dtin_bf, bc, nullptr, MB_SEQ, MB_XDBLN, MB_DINNER);

  // dtT = softplus(dt_in @ W_dt^T + b_dt), stored transposed (d_inner, L)
  wmma_gemm_kernel<128, 128, 2, 4, 4, 2, 1>
      <<<dim3(MB_DINNER / 128, MB_SEQ / 128), 256, 0, stream>>>(
          dtin_bf, Wdt_bf, dtT, nullptr, nullptr, b_dt, MB_SEQ, MB_DINNER, MB_DTRANK);

  // selective scan (includes + D*xs)
  scan_kernel<<<MB_DINNER / 16, 256, 0, stream>>>(dtT, xsT, bc, A_log, Dv, y);

  // gating by silu(z)
  gate_silu_kernel<<<(MB_SEQ * MB_DINNER) / 256, 256, 0, stream>>>(y, xz, yg_bf);

  // out = yg @ W_out^T
  wmma_gemm_kernel<128, 128, 2, 4, 4, 2, 0>
      <<<dim3(MB_DMODEL / 128, MB_SEQ / 128), 256, 0, stream>>>(
          yg_bf, Wout_bf, (float*)d_out, nullptr, nullptr, nullptr, MB_SEQ, MB_DMODEL, MB_DINNER);
}
